// AutoCorrelation_28467043238445
// MI455X (gfx1250) — matrix operand order repaired and checked by reading, not compiled, not run on hardware
//
#include <hip/hip_runtime.h>
#include <cstdint>

#define T_SEQ   3072
#define DMODEL  512
#define NHEAD   8
#define DH      64
#define NBATCH  16
#define KTOP    8
#define NROWS   (NBATCH * T_SEQ)   // 49152
#define NBH     (NBATCH * NHEAD)   // 128
#define NTT     24                 // t-tiles of 128 per (b,h)

typedef __attribute__((ext_vector_type(16))) __bf16 v16bf;
typedef __attribute__((ext_vector_type(8)))  __bf16 v8bf;
typedef __attribute__((ext_vector_type(8)))  float  v8f;

// ---------------- helpers ----------------

__device__ __forceinline__ uint16_t f2bf(float x) {       // RNE fp32 -> bf16
  uint32_t u = __float_as_uint(x);
  u += 0x7FFFu + ((u >> 16) & 1u);
  return (uint16_t)(u >> 16);
}

__device__ __forceinline__ v16bf cat16(const uint16_t* p0, const uint16_t* p1) {
  v8bf a = *(const v8bf*)p0;
  v8bf b = *(const v8bf*)p1;
  return __builtin_shufflevector(a, b, 0,1,2,3,4,5,6,7,8,9,10,11,12,13,14,15);
}

// A fragment 16x32 bf16 (ISA 7.12.2): lane<16 -> M=lane, K halves [g*8..g*8+8) and [16+g*8..)
__device__ __forceinline__ v16bf load_a_frag(const uint16_t* base, int ldh, int lane, int kc) {
  const int g = lane >> 4, m = lane & 15;
  const uint16_t* p = base + m * ldh + kc + g * 8;
  return cat16(p, p + 16);
}
// B fragment 32x16 bf16 stored [N][K] in LDS: lane -> N=lane&15, K halves [g*8..g*8+8) and [16+g*8..)
__device__ __forceinline__ v16bf load_b_frag(const uint16_t* base, int ldh, int lane, int kc) {
  const int g = lane >> 4, n = lane & 15;
  const uint16_t* p = base + n * ldh + kc + g * 8;
  return cat16(p, p + 16);
}

// CDNA5 async global->LDS copy (VGLOBAL GLOBAL_LOAD_ASYNC_TO_LDS_B128, ASYNCcnt)
__device__ __forceinline__ void async_copy_b128(uint32_t lds_addr, const void* gptr) {
  asm volatile("global_load_async_to_lds_b128 %0, %1, off"
               :
               : "v"(lds_addr), "v"((uint64_t)(uintptr_t)gptr)
               : "memory");
}
__device__ __forceinline__ void wait_async0() {
  asm volatile("s_wait_asynccnt 0x0" ::: "memory");
}
__device__ __forceinline__ uint32_t lds_addr_of(const void* p) {
  return (uint32_t)(uintptr_t)p;   // low 32 bits of generic LDS pointer = wave-relative LDS addr
}

// ---------------- GEMM: C[M x 512] = A[M x 512] * W[512 x 512] + bias ----------------
// OUT_MODE 0: bf16, head-split layout [b][h][t][dh]    OUT_MODE 1: fp32 row-major
template <bool A_BF16, int OUT_MODE>
__global__ __launch_bounds__(256) void gemm512_wmma(const void* __restrict__ Ain,
                                                    const float* __restrict__ W,
                                                    const float* __restrict__ bias,
                                                    void* __restrict__ Out) {
  constexpr int LDH = 40;                       // halves per LDS row (32 + pad), 80B = 5x16B
  __shared__ uint16_t Al[128 * LDH];
  __shared__ uint16_t Bl[128 * LDH];
  const int tid  = threadIdx.x;
  const int lane = tid & 31;
  const int wv   = tid >> 5;
  const int wm   = wv >> 2;                     // 0..1  (wave grid 2 x 4)
  const int wn   = wv & 3;                      // 0..3
  const int m0   = blockIdx.y * 128;
  const int n0   = blockIdx.x * 128;

  v8f C[4][2];
  #pragma unroll
  for (int mi = 0; mi < 4; ++mi)
    #pragma unroll
    for (int ni = 0; ni < 2; ++ni)
      #pragma unroll
      for (int v = 0; v < 8; ++v) C[mi][ni][v] = 0.0f;

  for (int kc = 0; kc < DMODEL; kc += 32) {
    __syncthreads();
    if (A_BF16) {
      const uint16_t* Ag = (const uint16_t*)Ain;
      #pragma unroll
      for (int i = 0; i < 2; ++i) {
        int idx = tid + i * 256;
        int r = idx >> 2, q = idx & 3;
        *(uint4*)&Al[r * LDH + q * 8] =
            *(const uint4*)&Ag[(size_t)(m0 + r) * DMODEL + kc + q * 8];
      }
    } else {
      const float* Ag = (const float*)Ain;
      #pragma unroll
      for (int i = 0; i < 4; ++i) {
        int idx = tid + i * 256;
        int r = idx >> 3, q = idx & 7;
        float4 v = *(const float4*)&Ag[(size_t)(m0 + r) * DMODEL + kc + q * 4];
        uint2 h;
        h.x = (uint32_t)f2bf(v.x) | ((uint32_t)f2bf(v.y) << 16);
        h.y = (uint32_t)f2bf(v.z) | ((uint32_t)f2bf(v.w) << 16);
        *(uint2*)&Al[r * LDH + q * 4] = h;
      }
    }
    // B tile: W rows kc..kc+31, cols n0..n0+127, transposed into Bl[n][k]
    #pragma unroll
    for (int i = 0; i < 4; ++i) {
      int idx = tid + i * 256;
      int k = idx >> 5, q = idx & 31;
      float4 v = *(const float4*)&W[(size_t)(kc + k) * DMODEL + n0 + q * 4];
      Bl[(q * 4 + 0) * LDH + k] = f2bf(v.x);
      Bl[(q * 4 + 1) * LDH + k] = f2bf(v.y);
      Bl[(q * 4 + 2) * LDH + k] = f2bf(v.z);
      Bl[(q * 4 + 3) * LDH + k] = f2bf(v.w);
    }
    __syncthreads();

    v16bf a[4], b[2];
    #pragma unroll
    for (int mi = 0; mi < 4; ++mi)
      a[mi] = load_a_frag(&Al[(wm * 64 + mi * 16) * LDH], LDH, lane, 0);
    #pragma unroll
    for (int ni = 0; ni < 2; ++ni)
      b[ni] = load_b_frag(&Bl[(wn * 32 + ni * 16) * LDH], LDH, lane, 0);
    #pragma unroll
    for (int mi = 0; mi < 4; ++mi)
      #pragma unroll
      for (int ni = 0; ni < 2; ++ni)
        C[mi][ni] = __builtin_amdgcn_wmma_f32_16x16x32_bf16(
            false, a[mi], false, b[ni], (short)0, C[mi][ni], false, false);
  }

  // epilogue: C layout -> lane N=lane&15, M = v + 8*(lane>>4)
  #pragma unroll
  for (int mi = 0; mi < 4; ++mi) {
    #pragma unroll
    for (int ni = 0; ni < 2; ++ni) {
      #pragma unroll
      for (int v = 0; v < 8; ++v) {
        int row = m0 + wm * 64 + mi * 16 + ((lane >> 4) << 3) + v;
        int col = n0 + wn * 32 + ni * 16 + (lane & 15);
        float val = C[mi][ni][v] + bias[col];
        if (OUT_MODE == 0) {
          int bb = row / T_SEQ;
          int t  = row - bb * T_SEQ;
          int h  = col >> 6;
          int d  = col & 63;
          ((uint16_t*)Out)[(((size_t)bb * NHEAD + h) * T_SEQ + t) * DH + d] = f2bf(val);
        } else {
          ((float*)Out)[(size_t)row * DMODEL + col] = val;
        }
      }
    }
  }
}

// ---------------- correlation: circular-diagonal sums of G = Q K^T per (b,h) ----------------
__global__ __launch_bounds__(256) void corr_wmma(const uint16_t* __restrict__ Qh,
                                                 const uint16_t* __restrict__ Kh,
                                                 float* __restrict__ partial) {
  constexpr int LDH = 72;                       // 64 + pad, 144B rows
  __shared__ float    bins[T_SEQ];              // 12 KB
  __shared__ uint16_t Al[128 * LDH];            // 18 KB
  __shared__ uint16_t Bl[128 * LDH];            // 18 KB
  const int tid  = threadIdx.x;
  const int lane = tid & 31;
  const int wv   = tid >> 5;
  const int wm   = wv >> 2, wn = wv & 3;
  const int tt   = blockIdx.x;                  // t-tile
  const int bh   = blockIdx.y;                  // b*8 + h
  const int t0   = tt * 128;
  const uint16_t* Q = Qh + (size_t)bh * T_SEQ * DH;
  const uint16_t* K = Kh + (size_t)bh * T_SEQ * DH;

  // Q tile copied once via async global->LDS DMA (no VGPR round trip)
  #pragma unroll
  for (int i = 0; i < 4; ++i) {
    int idx = tid + i * 256;
    int r = idx >> 3, q = idx & 7;
    async_copy_b128(lds_addr_of(&Al[r * LDH + q * 8]),
                    &Q[(size_t)(t0 + r) * DH + q * 8]);
  }
  for (int i = tid; i < T_SEQ; i += 256) bins[i] = 0.0f;

  for (int s0 = 0; s0 < T_SEQ; s0 += 128) {
    __syncthreads();
    #pragma unroll
    for (int i = 0; i < 4; ++i) {
      int idx = tid + i * 256;
      int r = idx >> 3, q = idx & 7;
      async_copy_b128(lds_addr_of(&Bl[r * LDH + q * 8]),
                      &K[(size_t)(s0 + r) * DH + q * 8]);
      if (s0 + 128 < T_SEQ)
        __builtin_prefetch(&K[(size_t)(s0 + 128 + r) * DH + q * 8], 0, 0);
    }
    wait_async0();          // this wave's async LDS writes complete
    __syncthreads();        // all waves' writes visible

    v8f C[4][2];
    #pragma unroll
    for (int mi = 0; mi < 4; ++mi)
      #pragma unroll
      for (int ni = 0; ni < 2; ++ni)
        #pragma unroll
        for (int v = 0; v < 8; ++v) C[mi][ni][v] = 0.0f;

    #pragma unroll
    for (int kc = 0; kc < DH; kc += 32) {
      v16bf a[4], b[2];
      #pragma unroll
      for (int mi = 0; mi < 4; ++mi)
        a[mi] = load_a_frag(&Al[(wm * 64 + mi * 16) * LDH], LDH, lane, kc);
      #pragma unroll
      for (int ni = 0; ni < 2; ++ni)
        b[ni] = load_b_frag(&Bl[(wn * 32 + ni * 16) * LDH], LDH, lane, kc);
      #pragma unroll
      for (int mi = 0; mi < 4; ++mi)
        #pragma unroll
        for (int ni = 0; ni < 2; ++ni)
          C[mi][ni] = __builtin_amdgcn_wmma_f32_16x16x32_bf16(
              false, a[mi], false, b[ni], (short)0, C[mi][ni], false, false);
    }

    // bin G[t,s] into tau = (t - s) mod T
    #pragma unroll
    for (int mi = 0; mi < 4; ++mi) {
      #pragma unroll
      for (int ni = 0; ni < 2; ++ni) {
        #pragma unroll
        for (int v = 0; v < 8; ++v) {
          int t = t0 + wm * 64 + mi * 16 + ((lane >> 4) << 3) + v;
          int s = s0 + wn * 32 + ni * 16 + (lane & 15);
          int d = t - s;
          if (d < 0) d += T_SEQ;
          atomicAdd(&bins[d], C[mi][ni][v]);
        }
      }
    }
  }
  __syncthreads();
  float* P = partial + ((size_t)bh * NTT + tt) * T_SEQ;
  for (int i = tid; i < T_SEQ; i += 256) P[i] = bins[i];
}

// ---------------- reduce partials -> meanCorr (mean over dh folded in) ----------------
__global__ __launch_bounds__(256) void reduce_partial(const float* __restrict__ partial,
                                                      float* __restrict__ meanCorr) {
  int i = blockIdx.x * 256 + threadIdx.x;       // 0 .. NBH*T_SEQ-1
  int bh  = i / T_SEQ;
  int tau = i - bh * T_SEQ;
  float s = 0.0f;
  #pragma unroll
  for (int tt = 0; tt < NTT; ++tt)
    s += partial[((size_t)bh * NTT + tt) * T_SEQ + tau];
  meanCorr[i] = s * (1.0f / 64.0f);
}

// ---------------- top-8 (desc, jax tie-break = lowest index) + softmax ----------------
__global__ __launch_bounds__(128) void topk_softmax(const float* __restrict__ meanCorr,
                                                    float* __restrict__ weights,
                                                    int* __restrict__ lags) {
  __shared__ float sv[128 * KTOP];
  __shared__ int   si[128 * KTOP];
  __shared__ int   head[128];
  const int bh = blockIdx.x, tid = threadIdx.x;
  const float* mc = meanCorr + (size_t)bh * T_SEQ;

  float bv[KTOP]; int bi[KTOP];
  #pragma unroll
  for (int j = 0; j < KTOP; ++j) { bv[j] = -3.4e38f; bi[j] = 0x7fffffff; }
  for (int idx = tid; idx < T_SEQ; idx += 128) {
    float v = mc[idx];
    if (v > bv[KTOP - 1]) {
      int j = KTOP - 1;
      while (j > 0 && v > bv[j - 1]) { bv[j] = bv[j - 1]; bi[j] = bi[j - 1]; --j; }
      bv[j] = v; bi[j] = idx;
    }
  }
  #pragma unroll
  for (int j = 0; j < KTOP; ++j) { sv[tid * KTOP + j] = bv[j]; si[tid * KTOP + j] = bi[j]; }
  head[tid] = 0;
  __syncthreads();

  if (tid == 0) {
    float vals[KTOP]; int idxs[KTOP];
    for (int r = 0; r < KTOP; ++r) {
      float best = -3.4e38f; int bidx = 0x7fffffff; int bth = 0;
      for (int th = 0; th < 128; ++th) {
        int hp = head[th];
        if (hp < KTOP) {
          float v = sv[th * KTOP + hp];
          int  ix = si[th * KTOP + hp];
          if (v > best || (v == best && ix < bidx)) { best = v; bidx = ix; bth = th; }
        }
      }
      vals[r] = best; idxs[r] = bidx; head[bth]++;
    }
    float m = vals[0], s = 0.0f, e[KTOP];
    for (int r = 0; r < KTOP; ++r) { e[r] = expf(vals[r] - m); s += e[r]; }
    float inv = 1.0f / s;
    for (int r = 0; r < KTOP; ++r) {
      weights[bh * KTOP + r] = e[r] * inv;
      lags[bh * KTOP + r]    = idxs[r];
    }
  }
}

// ---------------- scalar mean lag per rank (exact in fp32: integer sums < 2^24) ----------------
__global__ void lag_mean(const int* __restrict__ lags, int* __restrict__ lagShift) {
  int i = threadIdx.x;
  if (i < KTOP) {
    float s = 0.0f;
    for (int bh = 0; bh < NBH; ++bh) s += (float)lags[bh * KTOP + i];
    lagShift[i] = (int)(s * (1.0f / 128.0f));   // trunc toward zero, all >= 0
  }
}

// ---------------- weighted circular gather of V, write bf16 [N x 512] ----------------
__global__ __launch_bounds__(256) void roll_accum(const uint16_t* __restrict__ Vh,
                                                  const float* __restrict__ weights,
                                                  const int* __restrict__ lagShift,
                                                  uint16_t* __restrict__ attnO) {
  size_t gid = (size_t)blockIdx.x * 256 + threadIdx.x;   // NBH*T_SEQ*8 threads
  int q = (int)(gid & 7);                                // 8 halves per thread
  size_t r = gid >> 3;
  int t  = (int)(r % T_SEQ);
  int bh = (int)(r / T_SEQ);

  float w[KTOP]; int lg[KTOP];
  #pragma unroll
  for (int i = 0; i < KTOP; ++i) { w[i] = weights[bh * KTOP + i]; lg[i] = lagShift[i]; }

  float acc[8];
  #pragma unroll
  for (int j = 0; j < 8; ++j) acc[j] = 0.0f;

  #pragma unroll
  for (int i = 0; i < KTOP; ++i) {
    int ts = t + lg[i];
    if (ts >= T_SEQ) ts -= T_SEQ;                        // roll(V,-lag)[t] = V[(t+lag)%T]
    uint4 u = *(const uint4*)&Vh[((size_t)bh * T_SEQ + ts) * DH + q * 8];
    uint32_t ww[4] = {u.x, u.y, u.z, u.w};
    #pragma unroll
    for (int j = 0; j < 4; ++j) {
      acc[2 * j]     += w[i] * __uint_as_float(ww[j] << 16);
      acc[2 * j + 1] += w[i] * __uint_as_float(ww[j] & 0xffff0000u);
    }
  }
  uint4 o;
  o.x = (uint32_t)f2bf(acc[0]) | ((uint32_t)f2bf(acc[1]) << 16);
  o.y = (uint32_t)f2bf(acc[2]) | ((uint32_t)f2bf(acc[3]) << 16);
  o.z = (uint32_t)f2bf(acc[4]) | ((uint32_t)f2bf(acc[5]) << 16);
  o.w = (uint32_t)f2bf(acc[6]) | ((uint32_t)f2bf(acc[7]) << 16);
  int b = bh >> 3, h = bh & 7;
  *(uint4*)&attnO[((size_t)(b * T_SEQ + t)) * DMODEL + h * DH + q * 8] = o;
}

// ---------------- host ----------------
extern "C" void kernel_launch(void* const* d_in, const int* in_sizes, int n_in,
                              void* d_out, int out_size, void* d_ws, size_t ws_size,
                              hipStream_t stream) {
  (void)in_sizes; (void)n_in; (void)out_size; (void)ws_size;
  const float* query = (const float*)d_in[0];
  const float* key_  = (const float*)d_in[1];
  const float* value = (const float*)d_in[2];
  const float* Wq = (const float*)d_in[3];
  const float* Wk = (const float*)d_in[4];
  const float* Wv = (const float*)d_in[5];
  const float* Wo = (const float*)d_in[6];
  const float* bq = (const float*)d_in[7];
  const float* bk = (const float*)d_in[8];
  const float* bv = (const float*)d_in[9];
  const float* bo = (const float*)d_in[10];

  char* ws = (char*)d_ws;
  size_t off = 0;
  auto wsAlloc = [&](size_t bytes) -> void* {
    void* p = ws + off;
    off += (bytes + 255) & ~(size_t)255;
    return p;
  };
  uint16_t* Qh = (uint16_t*)wsAlloc((size_t)NBH * T_SEQ * DH * 2);   // 50.3 MB
  uint16_t* Kh = (uint16_t*)wsAlloc((size_t)NBH * T_SEQ * DH * 2);   // 50.3 MB
  uint16_t* Vh = (uint16_t*)wsAlloc((size_t)NBH * T_SEQ * DH * 2);   // 50.3 MB
  float* partial  = (float*)wsAlloc((size_t)NBH * NTT * T_SEQ * 4);  // 37.7 MB
  float* meanCorr = (float*)wsAlloc((size_t)NBH * T_SEQ * 4);
  float* weights  = (float*)wsAlloc((size_t)NBH * KTOP * 4);
  int*   lags     = (int*)wsAlloc((size_t)NBH * KTOP * 4);
  int*   lagShift = (int*)wsAlloc((size_t)KTOP * 4);
  uint16_t* attnO = Qh;   // Qh dead after corr_wmma; reuse for attention output

  dim3 ggrid(DMODEL / 128, NROWS / 128);   // (4, 384)
  gemm512_wmma<false, 0><<<ggrid, 256, 0, stream>>>(query, Wq, bq, Qh);
  gemm512_wmma<false, 0><<<ggrid, 256, 0, stream>>>(key_,  Wk, bk, Kh);
  gemm512_wmma<false, 0><<<ggrid, 256, 0, stream>>>(value, Wv, bv, Vh);
  corr_wmma<<<dim3(NTT, NBH), 256, 0, stream>>>(Qh, Kh, partial);
  reduce_partial<<<(NBH * T_SEQ) / 256, 256, 0, stream>>>(partial, meanCorr);
  topk_softmax<<<NBH, 128, 0, stream>>>(meanCorr, weights, lags);
  lag_mean<<<1, 32, 0, stream>>>(lags, lagShift);
  roll_accum<<<(NBH * T_SEQ * 8) / 256, 256, 0, stream>>>(Vh, weights, lagShift, attnO);
  gemm512_wmma<true, 1><<<ggrid, 256, 0, stream>>>(attnO, Wo, bo, d_out);
}
